// CRABLayer_29652454212568
// MI455X (gfx1250) — matrix operand order repaired and checked by reading, not compiled, not run on hardware
//
#include <hip/hip_runtime.h>
#include <hip/hip_bf16.h>
#include <cstdint>
#include <cstddef>

// ---------------- problem constants ----------------
#define NB   4      // batch
#define NC   64     // channels
#define HW   192    // height == width
#define NL5  5      // num_lat
#define CSTR 40     // LDS channel stride (elements) -> 80B row stride, 16B aligned

typedef __attribute__((ext_vector_type(16))) __bf16 v16bf;
typedef __attribute__((ext_vector_type(8)))  float  v8f;

union FragBits { uint4 u[2]; v16bf v; };

#if __has_builtin(__builtin_amdgcn_sched_barrier)
#define SCHED_BARRIER() __builtin_amdgcn_sched_barrier(0)
#else
#define SCHED_BARRIER() asm volatile("" ::: "memory")
#endif

// =====================================================================
// 1) adaptive avg-pool to 3x3: gl[b][c][ky*3+kx] = mean of 64x64 block
// =====================================================================
__global__ __launch_bounds__(256) void crab_pool(const float* __restrict__ xin,
                                                 float* __restrict__ gl) {
    int id = blockIdx.x;                 // over 4*64*9
    int kx = id % 3;
    int ky = (id / 3) % 3;
    int c  = (id / 9) % NC;
    int b  = id / (9 * NC);
    const float* base = xin + ((((size_t)b * NC + c) * HW + ky * 64) * HW + kx * 64);
    float sum = 0.f;
    for (int i = threadIdx.x; i < 4096; i += 256) {
        int r = i >> 6, col = i & 63;
        sum += base[(size_t)r * HW + col];
    }
    for (int off = 16; off > 0; off >>= 1) sum += __shfl_down(sum, off, 32);
    __shared__ float red[8];
    int lane = threadIdx.x & 31, wv = threadIdx.x >> 5;
    if (lane == 0) red[wv] = sum;
    __syncthreads();
    if (threadIdx.x == 0) {
        float t = 0.f;
        for (int w = 0; w < 8; ++w) t += red[w];
        gl[(b * NC + c) * 9 + ky * 3 + kx] = t * (1.f / 4096.f);
    }
}

// =====================================================================
// 2) gating network: one block, all batches (BN stats span batches)
//    produces g1[b][c][9], g2[b][oc][9]
// =====================================================================
__global__ __launch_bounds__(256) void crab_gating(
    const float* __restrict__ gl,
    const float* __restrict__ wg,   const float* __restrict__ wr,
    const float* __restrict__ aff1, const float* __restrict__ aff2,
    const float* __restrict__ ce,   const float* __restrict__ gd,
    const float* __restrict__ gd2,  const float* __restrict__ ci,
    const float* __restrict__ ce_g, const float* __restrict__ ce_b,
    const float* __restrict__ bn2_g,const float* __restrict__ bn2_b,
    const float* __restrict__ bn_g, const float* __restrict__ bn_b,
    float* __restrict__ g1, float* __restrict__ g2) {

    __shared__ float s_out[NB * NC * NL5];
    __shared__ float s_A  [NB * NC * NL5];   // w1 / v (ci path)
    __shared__ float s_B  [NB * NC * NL5];   // w2 / relu(bn2(ce2))
    __shared__ float s_aff[NB * 25];
    __shared__ float s_C  [NB * NL5 * NC];   // out1
    __shared__ float s_D  [NB * NL5 * NC];   // tmp
    __shared__ float s_E  [NB * NL5 * NC];   // out3
    __shared__ float s_ce2[NB * NC * NL5];
    __shared__ float s_sc [NC], s_sh[NC];

    const int tid = threadIdx.x;

    // out = gl @ ce^T   (b,c,5)
    for (int i = tid; i < NB * NC * NL5; i += 256) {
        int j = i % NL5, c = (i / NL5) % NC, b = i / (NC * NL5);
        float a = 0.f;
        for (int k = 0; k < 9; ++k) a += gl[(b * NC + c) * 9 + k] * ce[j * 9 + k];
        s_out[(b * NC + c) * NL5 + j] = a;
    }
    __syncthreads();
    // w1 = aff1 @ out ; w2 = aff2 @ out
    for (int i = tid; i < NB * NC * NL5; i += 256) {
        int j = i % NL5, r = (i / NL5) % NC, b = i / (NC * NL5);
        float a1 = 0.f, a2 = 0.f;
        for (int c = 0; c < NC; ++c) {
            float o = s_out[(b * NC + c) * NL5 + j];
            a1 += aff1[r * NC + c] * o;
            a2 += aff2[r * NC + c] * o;
        }
        s_A[(b * NC + r) * NL5 + j] = a1;
        s_B[(b * NC + r) * NL5 + j] = a2;
    }
    __syncthreads();
    // affinity = w1^T @ w2  (b,5,5)
    for (int i = tid; i < NB * 25; i += 256) {
        int j = i % NL5, ii = (i / NL5) % NL5, b = i / 25;
        float a = 0.f;
        for (int c = 0; c < NC; ++c)
            a += s_A[(b * NC + c) * NL5 + ii] * s_B[(b * NC + c) * NL5 + j];
        s_aff[b * 25 + ii * NL5 + j] = a;
    }
    __syncthreads();
    // out1 = affinity @ out^T (b,5,c)
    for (int i = tid; i < NB * NL5 * NC; i += 256) {
        int c = i % NC, ii = (i / NC) % NL5, b = i / (NL5 * NC);
        float a = 0.f;
        for (int j = 0; j < NL5; ++j)
            a += s_aff[b * 25 + ii * NL5 + j] * s_out[(b * NC + c) * NL5 + j];
        s_C[(b * NL5 + ii) * NC + c] = a;
    }
    __syncthreads();
    // tmp = out1 @ wg
    for (int i = tid; i < NB * NL5 * NC; i += 256) {
        int c = i % NC, ii = (i / NC) % NL5, b = i / (NL5 * NC);
        float a = 0.f;
        for (int cp = 0; cp < NC; ++cp)
            a += s_C[(b * NL5 + ii) * NC + cp] * wg[cp * NC + c];
        s_D[(b * NL5 + ii) * NC + c] = a;
    }
    __syncthreads();
    // out3 = tmp @ wr
    for (int i = tid; i < NB * NL5 * NC; i += 256) {
        int c = i % NC, ii = (i / NC) % NL5, b = i / (NL5 * NC);
        float a = 0.f;
        for (int cp = 0; cp < NC; ++cp)
            a += s_D[(b * NL5 + ii) * NC + cp] * wr[cp * NC + c];
        s_E[(b * NL5 + ii) * NC + c] = a;
    }
    __syncthreads();
    // ce2 = out * (1 + out3^T)
    for (int i = tid; i < NB * NC * NL5; i += 256) {
        int j = i % NL5, c = (i / NL5) % NC, b = i / (NC * NL5);
        float o = s_out[(b * NC + c) * NL5 + j];
        s_ce2[(b * NC + c) * NL5 + j] = o * (1.f + s_E[(b * NL5 + j) * NC + c]);
    }
    __syncthreads();
    // ce_bn stats (batch+L), scale/shift
    if (tid < NC) {
        float m = 0.f, m2 = 0.f;
        for (int b = 0; b < NB; ++b)
            for (int j = 0; j < NL5; ++j) {
                float v = s_ce2[(b * NC + tid) * NL5 + j];
                m += v; m2 += v * v;
            }
        m *= 0.05f; m2 *= 0.05f;
        float sc = ce_g[tid] * rsqrtf(m2 - m * m + 1e-5f);
        s_sc[tid] = sc; s_sh[tid] = ce_b[tid] - m * sc;
    }
    __syncthreads();
    // gate1 = relu(bn(ce2)) @ gd^T   (tid covers exactly (b,c))
    {
        int c = tid % NC, b = tid / NC;
        float rb[NL5];
        for (int j = 0; j < NL5; ++j)
            rb[j] = fmaxf(s_ce2[(b * NC + c) * NL5 + j] * s_sc[c] + s_sh[c], 0.f);
        for (int kk = 0; kk < 9; ++kk) {
            float a = 0.f;
            for (int j = 0; j < NL5; ++j) a += rb[j] * gd[kk * NL5 + j];
            g1[(b * NC + c) * 9 + kk] = a;
        }
    }
    __syncthreads();
    // ci path: bn2 stats
    if (tid < NC) {
        float m = 0.f, m2 = 0.f;
        for (int b = 0; b < NB; ++b)
            for (int j = 0; j < NL5; ++j) {
                float v = s_ce2[(b * NC + tid) * NL5 + j];
                m += v; m2 += v * v;
            }
        m *= 0.05f; m2 *= 0.05f;
        float sc = bn2_g[tid] * rsqrtf(m2 - m * m + 1e-5f);
        s_sc[tid] = sc; s_sh[tid] = bn2_b[tid] - m * sc;
    }
    __syncthreads();
    // ru = relu(bn2(ce2))
    for (int i = tid; i < NB * NC * NL5; i += 256) {
        int j = i % NL5, c = (i / NL5) % NC, b = i / (NC * NL5);
        s_B[(b * NC + c) * NL5 + j] =
            fmaxf(s_ce2[(b * NC + c) * NL5 + j] * s_sc[c] + s_sh[c], 0.f);
    }
    __syncthreads();
    // group (16) mix: v[b][oc][j] = sum_gi ru[b][gblk*16+gi][j] * ci[go][gi]
    for (int i = tid; i < NB * NC * NL5; i += 256) {
        int j = i % NL5, oc = (i / NL5) % NC, b = i / (NC * NL5);
        int gblk = oc >> 4, go = oc & 15;
        float a = 0.f;
        for (int gi = 0; gi < 16; ++gi)
            a += s_B[(b * NC + gblk * 16 + gi) * NL5 + j] * ci[go * 16 + gi];
        s_A[(b * NC + oc) * NL5 + j] = a;
    }
    __syncthreads();
    // ci_bn stats on v
    if (tid < NC) {
        float m = 0.f, m2 = 0.f;
        for (int b = 0; b < NB; ++b)
            for (int j = 0; j < NL5; ++j) {
                float v = s_A[(b * NC + tid) * NL5 + j];
                m += v; m2 += v * v;
            }
        m *= 0.05f; m2 *= 0.05f;
        float sc = bn_g[tid] * rsqrtf(m2 - m * m + 1e-5f);
        s_sc[tid] = sc; s_sh[tid] = bn_b[tid] - m * sc;
    }
    __syncthreads();
    // gate2 = relu(bn(v)) @ gd2^T
    {
        int oc = tid % NC, b = tid / NC;
        float rb[NL5];
        for (int j = 0; j < NL5; ++j)
            rb[j] = fmaxf(s_A[(b * NC + oc) * NL5 + j] * s_sc[oc] + s_sh[oc], 0.f);
        for (int kk = 0; kk < 9; ++kk) {
            float a = 0.f;
            for (int j = 0; j < NL5; ++j) a += rb[j] * gd2[kk * NL5 + j];
            g2[(b * NC + oc) * 9 + kk] = a;
        }
    }
}

// =====================================================================
// 3) materialize dynamic kernel, bf16, layout [b][kk][oc][c]
// =====================================================================
__global__ __launch_bounds__(256) void crab_kernmat(const float* __restrict__ g1,
                                                    const float* __restrict__ g2,
                                                    const float* __restrict__ wconv,
                                                    __bf16* __restrict__ kern) {
    int i = blockIdx.x * 256 + threadIdx.x;      // over 4*9*64*64
    if (i >= NB * 9 * NC * NC) return;
    int c  = i % NC;
    int oc = (i / NC) % NC;
    int kk = (i / (NC * NC)) % 9;
    int b  = i / (NC * NC * 9);
    float g = g1[(b * NC + c) * 9 + kk] + g2[(b * NC + oc) * 9 + kk];
    float s = 1.f / (1.f + __expf(-g));
    kern[i] = (__bf16)(s * wconv[(oc * NC + c) * 9 + kk]);
}

// =====================================================================
// 4) implicit-GEMM conv: block = (b, py); M=64(oc) x N=192(px), K=576
//    K reordered as (kk, c); bf16 WMMA 16x16x32, fp32 accumulate.
//    Inner loop: all 6 B fragments loaded before a scheduling barrier so
//    the ds_load_b128s issue as one clause-able burst with partial waits,
//    then 6 back-to-back WMMAs; A-fragment global load software-pipelined.
//    layer==0: out = relu(y)   layer==1: out = y + xres
// =====================================================================
__global__ __launch_bounds__(256) void crab_conv(const float* __restrict__ xin,
                                                 const __bf16* __restrict__ kern,
                                                 const float* __restrict__ xres,
                                                 float* __restrict__ yout,
                                                 int layer) {
    __shared__ __attribute__((aligned(16))) __bf16 xs[3 * 194 * CSTR]; // 46,560 B

    const int py = blockIdx.x;         // 0..191
    const int b  = blockIdx.y;         // 0..3
    const int tid  = threadIdx.x;
    const int lane = tid & 31;
    const int wave = tid >> 5;
    const int wm = wave & 3;           // M tile (16 oc)
    const int wn = wave >> 2;          // N half (96 px)

    v8f acc[6] = {};

    const int ocr = wm * 16 + (lane & 15);
    const int kbA = (lane >> 4) * 8;   // A-fragment K base within 32
    const int kbB = (lane >> 4) * 8;   // B-fragment K base within 32

    // A-fragment pointer for linear chunk q = chalf*9 + kk
    auto aptr = [&](int q) -> const __bf16* {
        int ch = q / 9, kk = q % 9;
        return kern + ((((size_t)b * 9 + kk) * NC + ocr) * NC + ch * 32 + kbA);
    };

    // prefetch A for chunk 0
    FragBits fa;
    {
        const __bf16* ap = aptr(0);
        fa.u[0] = *reinterpret_cast<const uint4*>(ap);
        fa.u[1] = *reinterpret_cast<const uint4*>(ap + 16);
    }

    for (int chalf = 0; chalf < 2; ++chalf) {
        __syncthreads();
        // stage 3 halo-padded rows of 32 channels, fp32 -> bf16, [row_pos][c]
        for (int e = tid; e < 3 * 194 * 32; e += 256) {
            int p  = e % 194;               // px + 1 (halo)
            int c  = (e / 194) % 32;
            int dy = e / (194 * 32);
            int px = p - 1;
            int ry = py - 1 + dy;
            float v = 0.f;
            if (px >= 0 && px < HW && ry >= 0 && ry < HW)
                v = xin[(((size_t)b * NC + chalf * 32 + c) * HW + ry) * HW + px];
            xs[(dy * 194 + p) * CSTR + c] = (__bf16)v;
        }
        __syncthreads();

        for (int kk = 0; kk < 9; ++kk) {
            const int q = chalf * 9 + kk;

            // software-pipeline: issue next chunk's A global load now
            FragBits fnext = fa;
            if (q < 17) {
                const __bf16* ap = aptr(q + 1);
                fnext.u[0] = *reinterpret_cast<const uint4*>(ap);
                fnext.u[1] = *reinterpret_cast<const uint4*>(ap + 16);
            }

            // load ALL 6 B fragments first (12 ds_load_b128, clause-able)
            const int ky = kk / 3, kx = kk % 3;
            FragBits fb[6];
#pragma unroll
            for (int s = 0; s < 6; ++s) {
                int px = wn * 96 + s * 16 + (lane & 15);
                const __bf16* bp = &xs[(ky * 194 + px + kx) * CSTR + kbB];
                fb[s].u[0] = *reinterpret_cast<const uint4*>(bp);
                fb[s].u[1] = *reinterpret_cast<const uint4*>(bp + 16);
            }

            // keep all fragments live: no WMMA may be hoisted above, no load
            // may sink below — forces batched issue + partial dscnt waits
            SCHED_BARRIER();

            // 6 back-to-back WMMAs (independent accumulators, shared A)
#pragma unroll
            for (int s = 0; s < 6; ++s) {
                acc[s] = __builtin_amdgcn_wmma_f32_16x16x32_bf16(
                    false, fa.v, false, fb[s].v, (short)0, acc[s], false, false);
            }

            fa = fnext;
        }
    }

    // epilogue: C/D layout — lanes 0-15: M=r, N=lane; lanes 16-31: M=r+8
#pragma unroll
    for (int s = 0; s < 6; ++s) {
#pragma unroll
        for (int r = 0; r < 8; ++r) {
            int oc = wm * 16 + r + (lane >> 4) * 8;
            int px = wn * 96 + s * 16 + (lane & 15);
            size_t o = (((size_t)b * NC + oc) * HW + py) * HW + px;
            float v = acc[s][r];
            if (layer == 0) yout[o] = fmaxf(v, 0.f);
            else            yout[o] = v + xres[o];
        }
    }
}

// =====================================================================
// launch
// =====================================================================
extern "C" void kernel_launch(void* const* d_in, const int* in_sizes, int n_in,
                              void* d_out, int out_size, void* d_ws, size_t ws_size,
                              hipStream_t stream) {
    (void)in_sizes; (void)n_in; (void)out_size; (void)ws_size;
    // inputs: 0:x, then p1 leaves 1..15, p2 leaves 16..30 (dict order)
    const float* x = (const float*)d_in[0];
    auto P = [&](int set, int idx) -> const float* {
        return (const float*)d_in[1 + set * 15 + idx];
    };
    // leaf order: wg wr aff1 aff2 wconv ce gd gd2 ci ce_g ce_b bn2_g bn2_b bn_g bn_b

    char* ws = (char*)d_ws;
    const size_t TBYTES = (size_t)NB * NC * HW * HW * sizeof(float); // 37,748,736
    float*  t    = (float*)ws;
    float*  gl   = (float*)(ws + TBYTES);
    float*  g1   = (float*)(ws + TBYTES + 9216);
    float*  g2   = (float*)(ws + TBYTES + 2 * 9216);
    __bf16* kern = (__bf16*)(ws + TBYTES + 3 * 9216);
    float*  out  = (float*)d_out;

    dim3 cgrid(HW, NB);

    for (int layer = 0; layer < 2; ++layer) {
        const float* in = (layer == 0) ? x : t;
        int s = layer;  // param set
        crab_pool<<<NB * NC * 9, 256, 0, stream>>>(in, gl);
        crab_gating<<<1, 256, 0, stream>>>(gl,
            P(s,0), P(s,1), P(s,2), P(s,3), P(s,5), P(s,6), P(s,7), P(s,8),
            P(s,9), P(s,10), P(s,11), P(s,12), P(s,13), P(s,14), g1, g2);
        crab_kernmat<<<(NB * 9 * NC * NC) / 256, 256, 0, stream>>>(g1, g2, P(s,4), kern);
        if (layer == 0)
            crab_conv<<<cgrid, 256, 0, stream>>>(in, kern, x, t, 0);
        else
            crab_conv<<<cgrid, 256, 0, stream>>>(in, kern, x, out, 1);
    }
}
